// SparseConvNeXtV2_76716705841806
// MI455X (gfx1250) — matrix operand order repaired and checked by reading, not compiled, not run on hardware
//
#include <hip/hip_runtime.h>
#include <hip/hip_bf16.h>
#include <stdint.h>

// ---------------------------------------------------------------------------
// Types for CDNA5 WMMA (wave32): v_wmma_f32_16x16x32_bf16
// ---------------------------------------------------------------------------
typedef __attribute__((ext_vector_type(16))) __bf16          v16bf;
typedef __attribute__((ext_vector_type(8)))  float           v8f;
typedef __attribute__((ext_vector_type(8)))  unsigned short  us8;

__device__ __forceinline__ unsigned short f2bf(float f) {
  unsigned int u = __float_as_uint(f);
  u = u + 0x7fffu + ((u >> 16) & 1u);     // round-to-nearest-even
  return (unsigned short)(u >> 16);
}

__device__ __forceinline__ float gelu_tanh(float v) {
  float c = 0.7978845608028654f * (v + 0.044715f * v * v * v);
  return 0.5f * v * (1.0f + tanhf(c));
}

// mask value (1=active) for a flattened token row at resolution H=2^Hs,
// patch grid is always 4x4, s = Hs - 2.
__device__ __forceinline__ float site_mask(const int* __restrict__ mask,
                                           int row, int Hs, int s) {
  int H = 1 << Hs;
  int b = row >> (2 * Hs);
  int y = (row >> Hs) & (H - 1);
  int x = row & (H - 1);
  int pidx = ((y >> s) << 2) + (x >> s);
  return 1.0f - (float)mask[(b << 4) + pidx];
}

// ---------------------------------------------------------------------------
// fp32 -> bf16 weight conversion
// ---------------------------------------------------------------------------
__global__ void cvt_bf16_kernel(const float* __restrict__ S,
                                unsigned short* __restrict__ D, int n) {
  int i = blockIdx.x * 256 + threadIdx.x;
  if (i < n) D[i] = f2bf(S[i]);
}

// ---------------------------------------------------------------------------
// Fused: masked 3x3 conv (3->96) + LN + GELU + 4x4/s4 depthwise stem + LN.
// One workgroup (128 threads) per stem output token (b, Y, X) on the 32x32 grid.
// ---------------------------------------------------------------------------
__global__ void init_stem_kernel(const float* __restrict__ X,     // [32,3,128,128]
                                 const int* __restrict__ mask,    // [32,16]
                                 const float* __restrict__ cw,    // (3,3,3,96)
                                 const float* __restrict__ cb,    // (96)
                                 const float* __restrict__ lg,
                                 const float* __restrict__ lb,
                                 const float* __restrict__ sw,    // (4,4,1,96)
                                 const float* __restrict__ sb,
                                 const float* __restrict__ slg,
                                 const float* __restrict__ slb,
                                 float* __restrict__ O) {         // [32*32*32, 96]
  __shared__ float sv[16][96];
  __shared__ float smu[16], srs[16];
  __shared__ float sout[96];
  __shared__ float s2[2];
  int bid = blockIdx.x;
  int b   = bid >> 10;
  int rem = bid & 1023;
  int Yo = rem >> 5, Xo = rem & 31;
  float mval = 1.0f - (float)mask[(b << 4) + ((Yo >> 3) << 2) + (Xo >> 3)];
  int tid = threadIdx.x;  // 128 threads

  for (int e = tid; e < 16 * 96; e += 128) {
    int p = e / 96, c = e - p * 96;
    int py = p >> 2, px = p & 3;
    int iy0 = (Yo << 2) + py, ix0 = (Xo << 2) + px;
    float acc = 0.f;
    for (int dy = 0; dy < 3; ++dy) {
      int iy = iy0 + dy - 1;
      if ((unsigned)iy >= 128u) continue;
      for (int dx = 0; dx < 3; ++dx) {
        int ix = ix0 + dx - 1;
        if ((unsigned)ix >= 128u) continue;
        float im = 1.0f - (float)mask[(b << 4) + ((iy >> 5) << 2) + (ix >> 5)];
        for (int ci = 0; ci < 3; ++ci) {
          float xv = X[(((size_t)b * 3 + ci) << 14) + (iy << 7) + ix] * im;
          acc += xv * cw[((dy * 3 + dx) * 3 + ci) * 96 + c];
        }
      }
    }
    sv[p][c] = (acc + cb[c]) * mval;
  }
  __syncthreads();
  if (tid < 16) {
    float sum = 0.f, sq = 0.f;
    for (int c = 0; c < 96; ++c) { float v = sv[tid][c]; sum += v; sq += v * v; }
    float mu = sum * (1.f / 96.f);
    float var = sq * (1.f / 96.f) - mu * mu;
    smu[tid] = mu; srs[tid] = rsqrtf(var + 1e-6f);
  }
  __syncthreads();
  for (int e = tid; e < 16 * 96; e += 128) {
    int p = e / 96, c = e - p * 96;
    float v = (sv[p][c] - smu[p]) * srs[p] * lg[c] + lb[c];
    v *= mval;
    sv[p][c] = gelu_tanh(v);
  }
  __syncthreads();
  if (tid < 96) {
    float acc = 0.f;
    for (int p = 0; p < 16; ++p) acc += sv[p][tid] * sw[p * 96 + tid];
    sout[tid] = (acc + sb[tid]) * mval;
  }
  __syncthreads();
  if (tid == 0) {
    float sum = 0.f, sq = 0.f;
    for (int c = 0; c < 96; ++c) { float v = sout[c]; sum += v; sq += v * v; }
    float mu = sum * (1.f / 96.f);
    s2[0] = mu;
    s2[1] = rsqrtf(sq * (1.f / 96.f) - mu * mu + 1e-6f);
  }
  __syncthreads();
  if (tid < 96) {
    float v = (sout[tid] - s2[0]) * s2[1] * slg[tid] + slb[tid];
    v *= mval;
    O[((size_t)((b << 10) + (Yo << 5) + Xo)) * 96 + tid] = v;
  }
}

// ---------------------------------------------------------------------------
// 7x7 depthwise conv, SAME padding, + bias, * mask
// ---------------------------------------------------------------------------
__global__ void dwconv7_kernel(const float* __restrict__ X,
                               const float* __restrict__ W,   // (7,7,1,d)
                               const float* __restrict__ Bv,
                               const int* __restrict__ mask,
                               float* __restrict__ O,
                               int Hs, int d, int s, int total) {
  int idx = blockIdx.x * 256 + threadIdx.x;
  if (idx >= total) return;
  int row = idx / d, c = idx - row * d;
  int H = 1 << Hs;
  int b = row >> (2 * Hs);
  int y = (row >> Hs) & (H - 1);
  int x = row & (H - 1);
  float acc = 0.f;
  for (int dy = 0; dy < 7; ++dy) {
    int iy = y + dy - 3;
    if ((unsigned)iy >= (unsigned)H) continue;
    for (int dx = 0; dx < 7; ++dx) {
      int ix = x + dx - 3;
      if ((unsigned)ix >= (unsigned)H) continue;
      acc += X[(size_t)((b << (2 * Hs)) + (iy << Hs) + ix) * d + c] *
             W[(dy * 7 + dx) * d + c];
    }
  }
  float mval = site_mask(mask, row, Hs, s);
  O[idx] = (acc + Bv[c]) * mval;
}

// ---------------------------------------------------------------------------
// LayerNorm over channels, one wave32 per token. Writes fp32 in-place and
// bf16 copy (for the WMMA GEMM A operand).
// ---------------------------------------------------------------------------
__global__ void ln_kernel(float* __restrict__ T,
                          const float* __restrict__ g,
                          const float* __restrict__ bb,
                          const int* __restrict__ mask,
                          unsigned short* __restrict__ HB,
                          int M, int d, int Hs, int s) {
  int lane = threadIdx.x & 31;
  int token = blockIdx.x * 8 + (threadIdx.x >> 5);
  if (token >= M) return;
  float* trow = T + (size_t)token * d;
  float sum = 0.f, sq = 0.f;
  for (int c = lane; c < d; c += 32) {
    float v = trow[c];
    sum += v; sq += v * v;
  }
  for (int o = 16; o > 0; o >>= 1) {
    sum += __shfl_xor(sum, o, 32);
    sq  += __shfl_xor(sq,  o, 32);
  }
  float inv_d = 1.0f / (float)d;
  float mu = sum * inv_d;
  float var = sq * inv_d - mu * mu;
  float rs = rsqrtf(var + 1e-6f);
  float mval = site_mask(mask, token, Hs, s);
  unsigned short* hrow = HB + (size_t)token * d;
  for (int c = lane; c < d; c += 32) {
    float v = (trow[c] - mu) * rs * g[c] + bb[c];
    v *= mval;
    trow[c] = v;
    hrow[c] = f2bf(v);
  }
}

// ---------------------------------------------------------------------------
// WMMA bf16 GEMM core, templated on the number of 16-wide N sub-tiles so the
// K-loop is branch-free straight-line code: all B loads issue together, then
// TMAX back-to-back v_wmma_f32_16x16x32_bf16 (good XDL issue pattern).
//   A frag: lane L holds row m0+(L&15); lo 8 halfs at k0+8*(L>>4),
//           hi 8 halfs at k0+16+8*(L>>4)          (ISA 16-bit A 16x32 layout)
//   B frag: lane L holds col n0+(L&15); lo 8 at rows k0+8*(L>>4)+j, hi 8 at k0+16+8*(L>>4)+j
//   C/D:    vgpr r, lane L -> row m0 + r + 8*(L>>4), col n0 + (L&15)
// mode 0: gelu((acc+b)*m)   mode 1: (acc+b)*m + resid   mode 2: (acc+b)*m
// ---------------------------------------------------------------------------
union AU { struct { us8 lo, hi; } p; v16bf v; };

template <int TMAX>
__device__ __forceinline__ void gemm_core(const unsigned short* __restrict__ arow,
                                          const unsigned short* __restrict__ W,
                                          const float* __restrict__ bias,
                                          float* __restrict__ C,
                                          const float* __restrict__ resid,
                                          const int* __restrict__ mask,
                                          int K, int N, int Hs, int s, int mode,
                                          int mbase, int nbase, int lane) {
  v8f zero = {0.f, 0.f, 0.f, 0.f, 0.f, 0.f, 0.f, 0.f};
  v8f acc[TMAX];
#pragma unroll
  for (int t = 0; t < TMAX; ++t) acc[t] = zero;

  const unsigned short* wptr = W + (size_t)(8 * (lane >> 4)) * N + nbase + (lane & 15);
  const size_t wstep = (size_t)32 * N;

  for (int k0 = 0; k0 < K; k0 += 32) {
    AU a;
    a.p.lo = *reinterpret_cast<const us8*>(arow + k0);
    a.p.hi = *reinterpret_cast<const us8*>(arow + k0 + 16);
    if (k0 + 32 < K) __builtin_prefetch(arow + k0 + 32, 0, 3);
    AU b[TMAX];
#pragma unroll
    for (int t = 0; t < TMAX; ++t) {
      for (int j = 0; j < 8; ++j) {
        b[t].p.lo[j] = wptr[(size_t)j * N + t * 16];
        b[t].p.hi[j] = wptr[(size_t)(16 + j) * N + t * 16];
      }
    }
#pragma unroll
    for (int t = 0; t < TMAX; ++t)
      acc[t] = __builtin_amdgcn_wmma_f32_16x16x32_bf16(
          false, a.v, false, b[t].v, (short)0, acc[t], false, false);
    wptr += wstep;
  }

  const int colofs = lane & 15;
  const int rbase = mbase + ((lane >> 4) << 3);
  float mrow[8];
#pragma unroll
  for (int r = 0; r < 8; ++r) mrow[r] = site_mask(mask, rbase + r, Hs, s);

#pragma unroll
  for (int t = 0; t < TMAX; ++t) {
    int col = nbase + t * 16 + colofs;
    float bv = bias[col];
#pragma unroll
    for (int r = 0; r < 8; ++r) {
      float v = (acc[t][r] + bv) * mrow[r];
      size_t o = (size_t)(rbase + r) * N + col;
      if (mode == 0)      C[o] = gelu_tanh(v);
      else if (mode == 1) C[o] = v + resid[o];
      else                C[o] = v;
    }
  }
}

__global__ void gemm_bf16_wmma(const unsigned short* __restrict__ A,
                               const unsigned short* __restrict__ W,
                               const float* __restrict__ bias,
                               float* __restrict__ C,
                               const float* __restrict__ resid,
                               const int* __restrict__ mask,
                               int M, int K, int N, int Hs, int s, int mode) {
  const int lane = threadIdx.x & 31;
  const int wave = threadIdx.x >> 5;
  const int mbase = blockIdx.x * 64 + wave * 16;
  const int nbase = blockIdx.y * 64;
  int tmax = (N - nbase) >> 4;
  if (tmax > 4) tmax = 4;

  const unsigned short* arow =
      A + (size_t)(mbase + (lane & 15)) * K + 8 * (lane >> 4);

  switch (tmax) {   // wave-uniform dispatch; each core is a branch-free loop
    case 4: gemm_core<4>(arow, W, bias, C, resid, mask, K, N, Hs, s, mode, mbase, nbase, lane); break;
    case 3: gemm_core<3>(arow, W, bias, C, resid, mask, K, N, Hs, s, mode, mbase, nbase, lane); break;
    case 2: gemm_core<2>(arow, W, bias, C, resid, mask, K, N, Hs, s, mode, mbase, nbase, lane); break;
    default: gemm_core<1>(arow, W, bias, C, resid, mask, K, N, Hs, s, mode, mbase, nbase, lane); break;
  }
}

// ---------------------------------------------------------------------------
// GRN part 1: per-channel sum of squares over all tokens (one block / channel)
// ---------------------------------------------------------------------------
__global__ void colsumsq_kernel(const float* __restrict__ Y,
                                float* __restrict__ out, int M, int N) {
  int c = blockIdx.x;
  __shared__ float red[256];
  float sum = 0.f;
  for (int r = threadIdx.x; r < M; r += 256) {
    float v = Y[(size_t)r * N + c];
    sum += v * v;
  }
  red[threadIdx.x] = sum;
  __syncthreads();
  for (int st = 128; st > 0; st >>= 1) {
    if (threadIdx.x < st) red[threadIdx.x] += red[threadIdx.x + st];
    __syncthreads();
  }
  if (threadIdx.x == 0) out[c] = red[0];
}

// GRN part 2: Gx = sqrt(sumsq); Nx = Gx / (mean(Gx) + 1e-6)   (single block)
__global__ void grn_nx_kernel(const float* __restrict__ sumsq,
                              float* __restrict__ nx, int Cn) {
  __shared__ float gx[3072];
  __shared__ float red[256];
  for (int c = threadIdx.x; c < Cn; c += 256) gx[c] = sqrtf(sumsq[c]);
  __syncthreads();
  float sum = 0.f;
  for (int c = threadIdx.x; c < Cn; c += 256) sum += gx[c];
  red[threadIdx.x] = sum;
  __syncthreads();
  for (int st = 128; st > 0; st >>= 1) {
    if (threadIdx.x < st) red[threadIdx.x] += red[threadIdx.x + st];
    __syncthreads();
  }
  float mean = red[0] / (float)Cn;
  for (int c = threadIdx.x; c < Cn; c += 256)
    nx[c] = gx[c] / (mean + 1e-6f);
}

// GRN part 3: z = (g*y*Nx + b + y) * m  -> bf16 (A operand of pw2)
__global__ void grn_apply_kernel(const float* __restrict__ Y,
                                 const float* __restrict__ nx,
                                 const float* __restrict__ gg,
                                 const float* __restrict__ gb,
                                 const int* __restrict__ mask,
                                 unsigned short* __restrict__ Z,
                                 int Cn, int Hs, int s, int total) {
  int idx = blockIdx.x * 256 + threadIdx.x;
  if (idx >= total) return;
  int row = idx / Cn, c = idx - row * Cn;
  float v = Y[idx];
  float mval = site_mask(mask, row, Hs, s);
  float o = (gg[c] * v * nx[c] + gb[c] + v) * mval;
  Z[idx] = f2bf(o);
}

// ---------------------------------------------------------------------------
// 2x2/s2 im2col (fp32 -> bf16) so the downsample conv runs as a WMMA GEMM.
// out[tok, q*d + ci], q = dy*2+dx  (matches HWIO (2,2,Ci,Co) weight flatten)
// ---------------------------------------------------------------------------
__global__ void im2col2x2_kernel(const float* __restrict__ X,
                                 unsigned short* __restrict__ O,
                                 int Hn, int Hsn, int d, int total) {
  int idx = blockIdx.x * 256 + threadIdx.x;
  if (idx >= total) return;
  int Kc = 4 * d;
  int tok = idx / Kc, rem = idx - tok * Kc;
  int q = rem / d, ci = rem - q * d;
  int b = tok >> (2 * Hsn);
  int Y = (tok >> Hsn) & (Hn - 1);
  int Xo = tok & (Hn - 1);
  int H = Hn << 1;
  int iy = (Y << 1) + (q >> 1), ix = (Xo << 1) + (q & 1);
  float v = X[(size_t)((b << (2 * (Hsn + 1))) + iy * H + ix) * d + ci];
  O[idx] = f2bf(v);
}

// ---------------------------------------------------------------------------
// NHWC [32,4,4,768] -> NCHW [32,768,4,4]
// ---------------------------------------------------------------------------
__global__ void final_nchw_kernel(const float* __restrict__ X,
                                  float* __restrict__ O, int total) {
  int idx = blockIdx.x * 256 + threadIdx.x;
  if (idx >= total) return;
  int b = idx / (768 * 16);
  int rem = idx - b * 768 * 16;
  int c = rem / 16;
  int yx = rem - c * 16;
  O[idx] = X[(size_t)(b * 16 + yx) * 768 + c];
}

// ---------------------------------------------------------------------------
// Host orchestration. Param leaves follow jax pytree order (dict keys sorted):
//   0:x 1:mask | 2..13: down[i]{b,ln_b,ln_g,w} | 14..17: init{b,ln_b,ln_g,w}
//   18..47: stage0, 48..77: stage1, 78..167: stage2, 168..197: stage3
//     per block: {dw_b,dw_w,grn_b,grn_g,ln_b,ln_g,pw1_b,pw1_w,pw2_b,pw2_w}
//   198..201: stem{b,ln_b,ln_g,w}
// ---------------------------------------------------------------------------
extern "C" void kernel_launch(void* const* d_in, const int* in_sizes, int n_in,
                              void* d_out, int out_size, void* d_ws, size_t ws_size,
                              hipStream_t stream) {
  (void)in_sizes; (void)n_in; (void)out_size; (void)ws_size;
  const float* x_in = (const float*)d_in[0];
  const int* mask = (const int*)d_in[1];
  auto P = [&](int i) { return (const float*)d_in[i]; };

  static const int DIMS[4]   = {96, 192, 384, 768};
  static const int DEPTHS[4] = {3, 3, 9, 3};
  static const int STG[4]    = {18, 48, 78, 168};

  char* ws = (char*)d_ws;
  size_t off = 0;
  auto carve = [&](size_t bytes) -> char* {
    char* p = ws + off;
    off = (off + bytes + 255) & ~(size_t)255;
    return p;
  };
  float*          XBUF  = (float*)carve(3145728ULL * sizeof(float));   // x  [N,d]
  float*          T1    = (float*)carve(3145728ULL * sizeof(float));   // dwconv/LN tmp
  unsigned short* HBF   = (unsigned short*)carve(3145728ULL * 2);      // bf16 A operand
  float*          YBUF  = (float*)carve(12582912ULL * sizeof(float));  // hidden [N,4d]
  unsigned short* ZBF   = (unsigned short*)carve(12582912ULL * 2);     // bf16 A for pw2
  float*          SUMSQ = (float*)carve(3072 * sizeof(float));
  float*          NXB   = (float*)carve(3072 * sizeof(float));
  unsigned short* WBF   = (unsigned short*)carve(27426816ULL * 2);     // bf16 weights

  // --- convert GEMM weights to bf16 ---
  size_t w1off[4][9], w2off[4][9], dnoff[3];
  size_t woff = 0;
  for (int st = 0; st < 4; ++st) {
    int d = DIMS[st];
    int n1 = d * 4 * d;
    for (int blk = 0; blk < DEPTHS[st]; ++blk) {
      int base = STG[st] + blk * 10;
      w1off[st][blk] = woff;
      cvt_bf16_kernel<<<(n1 + 255) / 256, 256, 0, stream>>>(P(base + 7), WBF + woff, n1);
      woff += n1;
      w2off[st][blk] = woff;
      cvt_bf16_kernel<<<(n1 + 255) / 256, 256, 0, stream>>>(P(base + 9), WBF + woff, n1);
      woff += n1;
    }
  }
  for (int i = 0; i < 3; ++i) {
    int n = 4 * DIMS[i] * DIMS[i + 1];
    dnoff[i] = woff;
    cvt_bf16_kernel<<<(n + 255) / 256, 256, 0, stream>>>(P(2 + 4 * i + 3), WBF + woff, n);
    woff += n;
  }

  // --- init conv + stem (fused) -> XBUF [32*32*32, 96] ---
  init_stem_kernel<<<32 * 32 * 32, 128, 0, stream>>>(
      x_in, mask, P(17), P(14), P(16), P(15), P(201), P(198), P(200), P(199), XBUF);

  int H = 32, Hs = 5, s = 3;
  for (int st = 0; st < 4; ++st) {
    int d = DIMS[st];
    int M = 32 * H * H;
    int C4 = 4 * d;
    for (int blk = 0; blk < DEPTHS[st]; ++blk) {
      int base = STG[st] + blk * 10;
      dwconv7_kernel<<<(M * d + 255) / 256, 256, 0, stream>>>(
          XBUF, P(base + 1), P(base + 0), mask, T1, Hs, d, s, M * d);
      ln_kernel<<<(M + 7) / 8, 256, 0, stream>>>(
          T1, P(base + 5), P(base + 4), mask, HBF, M, d, Hs, s);
      gemm_bf16_wmma<<<dim3(M / 64, (C4 + 63) / 64), 128, 0, stream>>>(
          HBF, WBF + w1off[st][blk], P(base + 6), YBUF, nullptr, mask,
          M, d, C4, Hs, s, 0);
      colsumsq_kernel<<<C4, 256, 0, stream>>>(YBUF, SUMSQ, M, C4);
      grn_nx_kernel<<<1, 256, 0, stream>>>(SUMSQ, NXB, C4);
      grn_apply_kernel<<<(M * C4 + 255) / 256, 256, 0, stream>>>(
          YBUF, NXB, P(base + 3), P(base + 2), mask, ZBF, C4, Hs, s, M * C4);
      gemm_bf16_wmma<<<dim3(M / 64, (d + 63) / 64), 128, 0, stream>>>(
          ZBF, WBF + w2off[st][blk], P(base + 8), XBUF, XBUF, mask,
          M, C4, d, Hs, s, 1);
    }
    if (st < 3) {
      int dbase = 2 + 4 * st;
      ln_kernel<<<(M + 7) / 8, 256, 0, stream>>>(
          XBUF, P(dbase + 2), P(dbase + 1), mask, HBF, M, d, Hs, s);
      int Hn = H >> 1, Hsn = Hs - 1, sn = s - 1;
      int Mn = 32 * Hn * Hn;
      int Kc = 4 * d;
      int Cout = DIMS[st + 1];
      im2col2x2_kernel<<<(Mn * Kc + 255) / 256, 256, 0, stream>>>(
          XBUF, HBF, Hn, Hsn, d, Mn * Kc);
      gemm_bf16_wmma<<<dim3(Mn / 64, (Cout + 63) / 64), 128, 0, stream>>>(
          HBF, WBF + dnoff[st], P(dbase + 0), XBUF, nullptr, mask,
          Mn, Kc, Cout, Hsn, sn, 2);
      H = Hn; Hs = Hsn; s = sn;
    }
  }

  final_nchw_kernel<<<(32 * 768 * 16 + 255) / 256, 256, 0, stream>>>(
      XBUF, (float*)d_out, 32 * 768 * 16);
}
